// ElasticityNet_44418551775696
// MI455X (gfx1250) — matrix operand order repaired and checked by reading, not compiled, not run on hardware
//
#include <hip/hip_runtime.h>

// ---------- types ----------
typedef __bf16 bf16_t;
typedef __bf16  v8bf  __attribute__((ext_vector_type(8)));
typedef __bf16  v16bf __attribute__((ext_vector_type(16)));
typedef float   v8f   __attribute__((ext_vector_type(8)));
typedef unsigned int u32x4 __attribute__((ext_vector_type(4)));
typedef unsigned int tdm_u4 __attribute__((ext_vector_type(4)));
typedef int          tdm_i4 __attribute__((ext_vector_type(4)));
typedef int          tdm_i8 __attribute__((ext_vector_type(8)));

#define NN        128     // neurons per layer
#define NLAYERS   15      // fc2..fc16
#define HS        136     // padded LDS row stride in halves (272 B, multiple of 16 B)
#define MTILE     128     // points per workgroup

#if __has_builtin(__builtin_amdgcn_tensor_load_to_lds) && \
    __has_builtin(__builtin_amdgcn_s_wait_tensorcnt)
#define USE_TDM 1
#endif

// ---------------------------------------------------------------------------
// Pre-pass: convert Ws (15 x 128 x 128 f32, row-major [l][k][n]) into bf16,
// TRANSPOSED per layer: wt[l][n][k] = Ws[l][k][n]. This makes the WMMA
// B-fragment (lane = output column, 16 consecutive K values) a contiguous
// per-lane LDS load.
// ---------------------------------------------------------------------------
__global__ void prep_weights(const float* __restrict__ Ws,
                             bf16_t* __restrict__ wt) {
    int idx = blockIdx.x * blockDim.x + threadIdx.x;
    if (idx >= NLAYERS * NN * NN) return;
    int l   = idx >> 14;       // / 16384
    int rem = idx & 16383;
    int n   = rem >> 7;        // output column
    int k   = rem & 127;       // input row
    wt[idx] = (bf16_t)Ws[(l << 14) + (k << 7) + n];
}

#ifdef USE_TDM
// ---------------------------------------------------------------------------
// TDM: DMA one 128x128 bf16 weight tile (row-major, stride 128) from global
// into LDS with 16 B padding inserted after every 256 B row (-> HS=136 halves).
// Issued by a single wave; tracked with TENSORcnt.
//   D# group0: count=1 | lds_addr | global_addr(57b) | type=2
//   D# group1: data_size=2B, pad_enable, pad_interval=64dw, pad_amount=4dw,
//              tensor_dim0=128, tensor_dim1=128, tile_dim0=128, tile_dim1=128,
//              tensor_dim0_stride=128
//   groups 2/3: zero (2-D tensor; tile_dim2=0 marks unused)
// ---------------------------------------------------------------------------
__device__ __forceinline__ void tdm_load_weights(const bf16_t* gsrc, bf16_t* ldst) {
    unsigned long long ga = (unsigned long long)gsrc;
    unsigned lds_off = (unsigned)(unsigned long long)ldst; // low 32b = LDS byte addr

    tdm_u4 g0;
    g0[0] = 1u;                                            // count=1 (valid)
    g0[1] = lds_off;
    g0[2] = (unsigned)(ga & 0xFFFFFFFFull);
    g0[3] = (unsigned)((ga >> 32) & 0x01FFFFFFull) | (2u << 30); // type=2

    tdm_i8 g1;
    g1[0] = (int)((1u << 16)      // data_size = 2 bytes
                | (1u << 20)      // pad_enable
                | (5u << 22)      // pad_interval: 64 DWORDs (256 B row)
                | (3u << 25));    // pad_amount:   4 DWORDs (16 B)
    g1[1] = (int)(128u << 16);    // tensor_dim0 = 128 (low 16 -> bits[79:64..])
    g1[2] = (int)(128u << 16);    // tensor_dim1 = 128
    g1[3] = (int)(128u << 16);    // tile_dim0   = 128
    g1[4] = 128;                  // tile_dim1   = 128, tile_dim2 = 0
    g1[5] = 128;                  // tensor_dim0_stride = 128
    g1[6] = 0;                    // dim0_stride hi / dim1_stride lo
    g1[7] = 0;

    tdm_i4 z4 = {0, 0, 0, 0};
#if defined(__clang_major__) && __clang_major__ >= 23
    tdm_i8 z8 = {0, 0, 0, 0, 0, 0, 0, 0};
    __builtin_amdgcn_tensor_load_to_lds(g0, g1, z4, z4, z8, 0);
#else
    __builtin_amdgcn_tensor_load_to_lds(g0, g1, z4, z4, 0);
#endif
}
#endif // USE_TDM

// ---------------------------------------------------------------------------
// Fused 17-layer MLP. One workgroup = 128 points, 8 waves (wave32).
// Wave w computes a 32-row x 64-col block: mg = w>>1 (rows mg*32..),
// ng = w&1 (cols ng*64..). acc = 2x4 tiles of 16x16 f32.
// ---------------------------------------------------------------------------
__global__ __launch_bounds__(256) void mlp_fused(
    const float* __restrict__ x,    // [N,2]
    const float* __restrict__ W1,   // [2,128]
    const float* __restrict__ b1,   // [128]
    const float* __restrict__ bs,   // [15,128]
    const float* __restrict__ W17,  // [128,1]
    const float* __restrict__ b17,  // [1]
    const bf16_t* __restrict__ wt,  // [15,128,128] bf16 transposed
    float* __restrict__ out)        // [N,1]
{
    extern __shared__ char smem_raw[];
    bf16_t* hbuf0   = (bf16_t*)smem_raw;              // 128*HS halves
    bf16_t* hbuf1   = hbuf0 + NN * HS;
    bf16_t* wbuf0   = hbuf1 + NN * HS;
    bf16_t* wbuf1   = wbuf0 + NN * HS;
    float*  biasAll = (float*)(wbuf1 + NN * HS);      // 15*128 f32
    float*  w1s     = biasAll + NLAYERS * NN;         // 256 f32
    float*  b1s     = w1s + 2 * NN;                   // 128 f32
    float*  w17s    = b1s + NN;                       // 128 f32

    const int t    = threadIdx.x;
    const int wv   = t >> 5;
    const int lane = t & 31;
    const int lH   = lane >> 4;     // 0: lanes 0-15, 1: lanes 16-31
    const int lc   = lane & 15;     // column-within-tile / row-within-tile
    const int mg   = wv >> 1;       // row group  (0..3) -> rows mg*32
    const int ng   = wv & 1;        // col group  (0..1) -> cols ng*64
    const int row0 = blockIdx.x * MTILE;

    // ---- stage constants + layer-0 weights into LDS ----
    for (int i = t; i < NLAYERS * NN; i += 256) biasAll[i] = bs[i];
    if (t < 2 * NN) w1s[t] = W1[t];
    if (t < NN) { b1s[t] = b1[t]; w17s[t] = W17[t]; }

#ifdef USE_TDM
    if (wv == 0) {
        tdm_load_weights(wt, wbuf0);
        __builtin_amdgcn_s_wait_tensorcnt(0);
    }
#else
    for (int c = t; c < 2048; c += 256) {              // 2048 chunks of 8 halves
        int r  = c >> 4;
        int ch = (c & 15) << 3;
        *(u32x4*)(wbuf0 + r * HS + ch) = *(const u32x4*)(wt + (r << 7) + ch);
    }
#endif
    __syncthreads();

    // ---- fc1: [N,2] @ [2,128] + b, relu ; VALU (K=2) -> hbuf0 bf16 ----
    {
        int r      = t >> 1;                  // 0..127
        int cbase  = (t & 1) << 6;            // 0 or 64
        float2 xv  = ((const float2*)x)[row0 + r];
        bf16_t* hr = hbuf0 + r * HS + cbase;
        #pragma unroll 8
        for (int c = 0; c < 64; ++c) {
            int cc  = cbase + c;
            float h = fmaxf(0.0f, xv.x * w1s[cc] + xv.y * w1s[NN + cc] + b1s[cc]);
            hr[c] = (bf16_t)h;
        }
    }

    // ---- 15 fused 128x128 Linear+ReLU layers via WMMA bf16 ----
    const bf16_t* hsrc = hbuf0;
    bf16_t*       hdst = hbuf1;

    for (int l = 0; l < NLAYERS; ++l) {
        __syncthreads();   // hsrc + current wbuf are ready; wbN readers done

        const bf16_t* wb  = (l & 1) ? wbuf1 : wbuf0;
        bf16_t*       wbN = (l & 1) ? wbuf0 : wbuf1;

#ifdef USE_TDM
        // async DMA next layer's weights into the other LDS buffer
        if (l < NLAYERS - 1 && wv == 0)
            tdm_load_weights(wt + (l + 1) * (NN * NN), wbN);
#else
        // prefetch next layer's weights (global -> regs), stored to LDS later
        u32x4 wreg[8];
        if (l < NLAYERS - 1) {
            const bf16_t* src = wt + (l + 1) * (NN * NN);
            #pragma unroll
            for (int i = 0; i < 8; ++i) {
                int c = t + i * 256;
                wreg[i] = *(const u32x4*)(src + c * 8);
            }
        }
#endif

        v8f acc[2][4] = {};

        #pragma unroll
        for (int kc = 0; kc < 4; ++kc) {
            // B fragments: wb[col][k], lane lH=0 -> K kc*32..+15, lH=1 -> +16..+31
            v16bf bfrag[4];
            #pragma unroll
            for (int ns = 0; ns < 4; ++ns) {
                const bf16_t* bp = wb + (ng * 64 + ns * 16 + lc) * HS
                                      + kc * 32 + lH * 16;
                v8bf blo = *(const v8bf*)bp;
                v8bf bhi = *(const v8bf*)(bp + 8);
                bfrag[ns] = __builtin_shufflevector(blo, bhi,
                    0,1,2,3,4,5,6,7,8,9,10,11,12,13,14,15);
            }
            // A fragments: hsrc[row][k], lanes 0-15: K 0-15, 16-31: K 16-31
            #pragma unroll
            for (int ms = 0; ms < 2; ++ms) {
                const bf16_t* ap = hsrc + (mg * 32 + ms * 16 + lc) * HS
                                        + kc * 32 + lH * 16;
                v8bf alo = *(const v8bf*)ap;
                v8bf ahi = *(const v8bf*)(ap + 8);
                v16bf afrag = __builtin_shufflevector(alo, ahi,
                    0,1,2,3,4,5,6,7,8,9,10,11,12,13,14,15);
                #pragma unroll
                for (int ns = 0; ns < 4; ++ns) {
                    acc[ms][ns] = __builtin_amdgcn_wmma_f32_16x16x32_bf16(
                        false, afrag, false, bfrag[ns],
                        (short)0, acc[ms][ns], false, false);
                }
            }
        }

#ifndef USE_TDM
        // stash prefetched weights into the other LDS weight buffer
        if (l < NLAYERS - 1) {
            #pragma unroll
            for (int i = 0; i < 8; ++i) {
                int c  = t + i * 256;
                int r  = c >> 4;
                int ch = (c & 15) << 3;
                *(u32x4*)(wbN + r * HS + ch) = wreg[i];
            }
        }
#endif

        // bias + relu + bf16 convert -> hdst
        const float* bias = biasAll + l * NN;
        #pragma unroll
        for (int ms = 0; ms < 2; ++ms) {
            #pragma unroll
            for (int ns = 0; ns < 4; ++ns) {
                int cc = ng * 64 + ns * 16 + lc;
                float bv = bias[cc];
                #pragma unroll
                for (int v = 0; v < 8; ++v) {
                    int rr = mg * 32 + ms * 16 + v + lH * 8;
                    float val = fmaxf(0.0f, acc[ms][ns][v] + bv);
                    hdst[rr * HS + cc] = (bf16_t)val;
                }
            }
        }

#ifdef USE_TDM
        // only the issuing wave tracks TENSORcnt; ensure DMA landed before
        // the barrier that publishes wbN to the other waves
        if (wv == 0)
            __builtin_amdgcn_s_wait_tensorcnt(0);
#endif

        // swap activation buffers
        const bf16_t* tmp = hsrc; hsrc = hdst; hdst = (bf16_t*)tmp;
    }

    __syncthreads();

    // ---- fc17: per-point 128-dot-product + b17 ----
    if (t < NN) {
        const bf16_t* hr = hsrc + t * HS;
        float s = 0.0f;
        #pragma unroll 8
        for (int k = 0; k < NN; ++k) s += (float)hr[k] * w17s[k];
        out[row0 + t] = s + b17[0];
    }
}

// ---------------------------------------------------------------------------
extern "C" void kernel_launch(void* const* d_in, const int* in_sizes, int n_in,
                              void* d_out, int out_size, void* d_ws, size_t ws_size,
                              hipStream_t stream) {
    const float* x   = (const float*)d_in[0];
    const float* W1  = (const float*)d_in[1];
    const float* b1  = (const float*)d_in[2];
    const float* Ws  = (const float*)d_in[3];
    const float* bs  = (const float*)d_in[4];
    const float* W17 = (const float*)d_in[5];
    const float* b17 = (const float*)d_in[6];
    float* out = (float*)d_out;
    bf16_t* wt = (bf16_t*)d_ws;                 // 15*128*128 bf16 = 480 KB

    int n = in_sizes[0] / 2;                    // number of points (1048576)

    // 1) weight convert + transpose to bf16
    int welems = NLAYERS * NN * NN;
    prep_weights<<<(welems + 255) / 256, 256, 0, stream>>>(Ws, wt);

    // 2) fused MLP
    size_t smem = (size_t)(4 * NN * HS) * sizeof(bf16_t)        // 2x hbuf + 2x wbuf
                + (size_t)(NLAYERS * NN) * sizeof(float)        // biases
                + (size_t)(2 * NN + NN + NN) * sizeof(float);   // W1, b1, W17
    mlp_fused<<<n / MTILE, 256, smem, stream>>>(x, W1, b1, bs, W17, b17, wt, out);
}
